// CRNN_9002251453224
// MI455X (gfx1250) — matrix operand order repaired and checked by reading, not compiled, not run on hardware
//
#include <hip/hip_runtime.h>
#include <hip/hip_bf16.h>
#include <math.h>

// ---------------------------------------------------------------------------
// Types for CDNA5 WMMA (wave32). v16bf = A/B fragment (16x32 / 32x16 bf16),
// v8f = 16x16 f32 C/D fragment.
// ---------------------------------------------------------------------------
typedef __bf16 v16bf __attribute__((ext_vector_type(16)));
typedef float  v8f   __attribute__((ext_vector_type(8)));

__device__ __forceinline__ v8f wmma_bf16(v16bf a, v16bf b, v8f c) {
    return __builtin_amdgcn_wmma_f32_16x16x32_bf16(
        /*neg_a=*/false, a, /*neg_b=*/false, b,
        /*c_mod=*/(short)0, c, /*reuse_a=*/false, /*reuse_b=*/false);
}

// A fragment (16x32 bf16, M x K) from a row-major [M][stride] bf16 tile.
// Lane l<16 holds row l, K {0..7, 16..23}; lane l+16 holds row l, K {8..15, 24..31}.
__device__ __forceinline__ v16bf frag_a(const __bf16* tile, int stride) {
    int l = threadIdx.x & 31;
    const __bf16* p = tile + (l & 15) * stride + ((l >> 4) << 3);
    union { v16bf v; uint4 q[2]; } f;
    f.q[0] = *(const uint4*)(p);
    f.q[1] = *(const uint4*)(p + 16);
    return f.v;
}

// B fragment (32x16 bf16, K x N) from an [N][stride] (k-contiguous) bf16 matrix.
// Lane l<16 holds col l, K {0..7, 16..23}; lane l+16 holds col l, K {8..15, 24..31}.
__device__ __forceinline__ v16bf frag_b(const __bf16* mat, int stride, int colBase, int kk) {
    int l = threadIdx.x & 31;
    const __bf16* p = mat + (size_t)(colBase + (l & 15)) * stride + kk + ((l >> 4) << 3);
    union { v16bf v; uint4 q[2]; } f;
    f.q[0] = *(const uint4*)(p);
    f.q[1] = *(const uint4*)(p + 16);
    return f.v;
}

// ---------------------------------------------------------------------------
// Weight prep
// ---------------------------------------------------------------------------
__global__ __launch_bounds__(256) void k_convert_bf16(const float* __restrict__ src,
                                                      __bf16* __restrict__ dst, int n) {
    int i = blockIdx.x * 256 + threadIdx.x;
    if (i < n) dst[i] = (__bf16)src[i];
}

__global__ __launch_bounds__(256) void k_reorder_convw(const float* __restrict__ w,
                                                       __bf16* __restrict__ dst) {
    // w: [256][256][5][5] (OIHW) -> dst[tap*65536 + co*256 + ci]
    int i = blockIdx.x * 256 + threadIdx.x;
    if (i < 25 * 65536) {
        int tap = i >> 16;
        int rem = i & 65535;
        int co  = rem >> 8;
        int ci  = rem & 255;
        dst[i] = (__bf16)w[(size_t)(co * 256 + ci) * 25 + tap];
    }
}

// ---------------------------------------------------------------------------
// conv1 (Cin=1) fused: conv5x5 + bias + BN + ReLU + pool5, out bf16 NHWC
// ---------------------------------------------------------------------------
__global__ __launch_bounds__(256) void k_conv1(const float* __restrict__ x,
                                               const float* __restrict__ w1,
                                               const float* __restrict__ b1,
                                               const float* __restrict__ g1,
                                               const float* __restrict__ bt1,
                                               const float* __restrict__ m1,
                                               const float* __restrict__ v1,
                                               __bf16* __restrict__ out) {
    int pix = blockIdx.x;          // ((n*512)+h)*8 + wo
    int wo  = pix & 7;
    int h   = (pix >> 3) & 511;
    int n   = pix >> 12;

    __shared__ float patch[5][12];
    int tid = threadIdx.x;
    if (tid < 45) {
        int r  = tid / 9, cc = tid % 9;
        int hh = h + r - 2;
        int ww = wo * 5 + cc - 2;
        float val = 0.f;
        if (hh >= 0 && hh < 512 && ww >= 0 && ww < 40)
            val = x[((size_t)n * 512 + hh) * 40 + ww];
        patch[r][cc] = val;
    }
    __syncthreads();

    int c = tid;
    float wgt[25];
#pragma unroll
    for (int i = 0; i < 25; ++i) wgt[i] = w1[c * 25 + i];
    float s  = g1[c] * rsqrtf(v1[c] + 1e-5f);
    float sh = (b1[c] - m1[c]) * s + bt1[c];

    float best = -INFINITY;
#pragma unroll
    for (int w = 0; w < 5; ++w) {
        float acc = 0.f;
#pragma unroll
        for (int r = 0; r < 5; ++r)
#pragma unroll
            for (int q = 0; q < 5; ++q)
                acc += patch[r][w + q] * wgt[r * 5 + q];
        best = fmaxf(best, acc);
    }
    float val = fmaxf(best * s + sh, 0.f);
    out[(size_t)pix * 256 + c] = (__bf16)val;
}

// ---------------------------------------------------------------------------
// Implicit-GEMM conv (Cin=Cout=256, 5x5, pad 2) fused with BN+ReLU+pool<POOL>.
// Block tile 128(M) x 128(N), 8 waves (each 32x64), double-buffered LDS,
// padded LDS strides (40 bf16/row) to avoid bank conflicts.
// ---------------------------------------------------------------------------
#define LDSPAD 40

template <int WIN, int POOL>
__global__ __launch_bounds__(256) void k_conv_gemm(const __bf16* __restrict__ in,
                                                   const __bf16* __restrict__ wr, // [25][256][256] (tap,O,I)
                                                   const float* __restrict__ bb,
                                                   const float* __restrict__ gg,
                                                   const float* __restrict__ bt,
                                                   const float* __restrict__ mm,
                                                   const float* __restrict__ vv,
                                                   __bf16* __restrict__ out) {
    constexpr int WO  = WIN / POOL;
    constexpr int NCH = 25 * 8;               // K chunks of 32

    int bm = (blockIdx.x >> 1) * 128;         // M tile base (pixels)
    int nB = (blockIdx.x & 1) * 128;          // N tile base (cout)

    __shared__ __align__(16) __bf16 As[2][128 * LDSPAD];
    __shared__ __align__(16) __bf16 Bs[2][128 * LDSPAD];

    int tid  = threadIdx.x;
    int lane = tid & 31;
    int wave = tid >> 5;
    int wm   = (wave >> 1) * 32;              // 0,32,64,96
    int wn   = (wave & 1) * 64;               // 0,64

    float scale[4], shift[4];
#pragma unroll
    for (int ni = 0; ni < 4; ++ni) {
        int col   = nB + wn + ni * 16 + (lane & 15);
        float s   = gg[col] * rsqrtf(vv[col] + 1e-5f);
        scale[ni] = s;
        shift[ni] = (bb[col] - mm[col]) * s + bt[col];
    }

    v8f acc[2][4] = {};

    // A staging: thread -> (pixel row tid>>1, 16-channel half)
    int arow  = tid >> 1;
    int apart = (tid & 1) * 16;
    int am    = bm + arow;
    int an    = am / (512 * WIN);
    int ah    = (am / WIN) % 512;
    int aw    = am % WIN;

    // B staging: thread -> (n-row tid>>1, 16-k half)
    int brow = tid >> 1;
    int bk0  = (tid & 1) * 16;

    auto stage = [&](int c, int buf) {
        int tap = c >> 3;
        int kc  = (c & 7) * 32;
        int dh  = tap / 5 - 2;
        int dw  = tap % 5 - 2;
        int hs  = ah + dh, ws = aw + dw;
        uint4 a0 = {0, 0, 0, 0}, a1 = {0, 0, 0, 0};
        if (hs >= 0 && hs < 512 && ws >= 0 && ws < WIN) {
            const __bf16* src = in + (((size_t)(an * 512 + hs) * WIN + ws) * 256) + kc + apart;
            a0 = *(const uint4*)(src);
            a1 = *(const uint4*)(src + 8);
        }
        *(uint4*)(As[buf] + arow * LDSPAD + apart)     = a0;
        *(uint4*)(As[buf] + arow * LDSPAD + apart + 8) = a1;
        const __bf16* bsrc = wr + (size_t)tap * 65536 + (size_t)(nB + brow) * 256 + kc + bk0;
        *(uint4*)(Bs[buf] + brow * LDSPAD + bk0)     = *(const uint4*)(bsrc);
        *(uint4*)(Bs[buf] + brow * LDSPAD + bk0 + 8) = *(const uint4*)(bsrc + 8);
        if ((c & 7) == 0 && tap < 24)
            __builtin_prefetch(bsrc + 65536, 0, 3);   // next tap's weight slice -> L2
    };

    stage(0, 0);
    __syncthreads();

    for (int c = 0; c < NCH; ++c) {
        int cur = c & 1;
        if (c + 1 < NCH) stage(c + 1, cur ^ 1);

        v16bf af0 = frag_a(As[cur] + (wm + 0) * LDSPAD, LDSPAD);
        v16bf af1 = frag_a(As[cur] + (wm + 16) * LDSPAD, LDSPAD);
#pragma unroll
        for (int ni = 0; ni < 4; ++ni) {
            v16bf bf   = frag_b(Bs[cur], LDSPAD, wn + ni * 16, 0);
            acc[0][ni] = wmma_bf16(af0, bf, acc[0][ni]);
            acc[1][ni] = wmma_bf16(af1, bf, acc[1][ni]);
        }
        __syncthreads();
    }

    // Epilogue: BN + ReLU + pool<POOL> along W, write bf16 NHWC
    int hi = lane >> 4;
#pragma unroll
    for (int mi = 0; mi < 2; ++mi) {
        int m_half = bm + wm + mi * 16 + 8 * hi;
#pragma unroll
        for (int ni = 0; ni < 4; ++ni) {
            float av[8];
#pragma unroll
            for (int i = 0; i < 8; ++i) av[i] = acc[mi][ni][i];
            int col = nB + wn + ni * 16 + (lane & 15);
#pragma unroll
            for (int j = 0; j < 8 / POOL; ++j) {
                float p = av[j * POOL];
#pragma unroll
                for (int q = 1; q < POOL; ++q) p = fmaxf(p, av[j * POOL + q]);
                float val = fmaxf(p * scale[ni] + shift[ni], 0.f);
                int mo    = m_half + j * POOL;
                int n_img = mo / (512 * WIN);
                int h     = (mo / WIN) % 512;
                int wo    = (mo % WIN) / POOL;
                out[(((size_t)(n_img * 512 + h) * WO) + wo) * 256 + col] = (__bf16)val;
            }
        }
    }
}

// ---------------------------------------------------------------------------
// gi_all = feats(8192x256) @ w_ih^T(256x768) + b_ih (fp32 out)
// Block tile 128x128, double-buffered.
// ---------------------------------------------------------------------------
__global__ __launch_bounds__(256) void k_gemm_gi(const __bf16* __restrict__ feats,
                                                 const __bf16* __restrict__ wih, // [768][256]
                                                 const float* __restrict__ bih,
                                                 float* __restrict__ gi) {
    int bm = (blockIdx.x / 6) * 128;
    int nB = (blockIdx.x % 6) * 128;

    __shared__ __align__(16) __bf16 As[2][128 * LDSPAD];
    __shared__ __align__(16) __bf16 Bs[2][128 * LDSPAD];

    int tid  = threadIdx.x;
    int lane = tid & 31;
    int wave = tid >> 5;
    int wm   = (wave >> 1) * 32;
    int wn   = (wave & 1) * 64;

    v8f acc[2][4] = {};

    int arow  = tid >> 1;
    int apart = (tid & 1) * 16;
    int brow  = tid >> 1;
    int bk0   = (tid & 1) * 16;

    auto stage = [&](int c, int buf) {
        int kc = c * 32;
        const __bf16* src = feats + (size_t)(bm + arow) * 256 + kc + apart;
        *(uint4*)(As[buf] + arow * LDSPAD + apart)     = *(const uint4*)(src);
        *(uint4*)(As[buf] + arow * LDSPAD + apart + 8) = *(const uint4*)(src + 8);
        const __bf16* bsrc = wih + (size_t)(nB + brow) * 256 + kc + bk0;
        *(uint4*)(Bs[buf] + brow * LDSPAD + bk0)     = *(const uint4*)(bsrc);
        *(uint4*)(Bs[buf] + brow * LDSPAD + bk0 + 8) = *(const uint4*)(bsrc + 8);
    };

    stage(0, 0);
    __syncthreads();

    for (int c = 0; c < 8; ++c) {
        int cur = c & 1;
        if (c + 1 < 8) stage(c + 1, cur ^ 1);
        v16bf af0 = frag_a(As[cur] + (wm + 0) * LDSPAD, LDSPAD);
        v16bf af1 = frag_a(As[cur] + (wm + 16) * LDSPAD, LDSPAD);
#pragma unroll
        for (int ni = 0; ni < 4; ++ni) {
            v16bf bf   = frag_b(Bs[cur], LDSPAD, wn + ni * 16, 0);
            acc[0][ni] = wmma_bf16(af0, bf, acc[0][ni]);
            acc[1][ni] = wmma_bf16(af1, bf, acc[1][ni]);
        }
        __syncthreads();
    }

    int hi = lane >> 4;
#pragma unroll
    for (int mi = 0; mi < 2; ++mi) {
        int mbase = bm + wm + mi * 16 + 8 * hi;
#pragma unroll
        for (int ni = 0; ni < 4; ++ni) {
            int col    = nB + wn + ni * 16 + (lane & 15);
            float bias = bih[col];
            float av[8];
#pragma unroll
            for (int i = 0; i < 8; ++i) av[i] = acc[mi][ni][i];
#pragma unroll
            for (int i = 0; i < 8; ++i)
                gi[(size_t)(mbase + i) * 768 + col] = av[i] + bias;
        }
    }
}

// ---------------------------------------------------------------------------
// GRU scan + classifier. Single workgroup (8 waves), 512 sequential steps.
// h kept in LDS with padded stride (264) to avoid 16-way bank conflicts.
// ---------------------------------------------------------------------------
#define HSTRIDE 264

__global__ __launch_bounds__(256) void k_gru(const float* __restrict__ gi,    // [16][512][768]
                                             const __bf16* __restrict__ whh,  // [768][256]
                                             const float* __restrict__ bhh,   // [768]
                                             const __bf16* __restrict__ wcls, // [16][256]
                                             const float* __restrict__ bcls,  // [16]
                                             float* __restrict__ outp) {      // [16][512][16]
    __shared__ __align__(16) __bf16 hb[16 * HSTRIDE];
    __shared__ float hf[16 * 256];
    __shared__ float gh[16 * 768];

    int tid  = threadIdx.x;
    int lane = tid & 31;
    int wave = tid >> 5;

    for (int i = tid; i < 16 * 256; i += 256) hf[i] = 0.f;
    for (int i = tid; i < 16 * HSTRIDE; i += 256) hb[i] = (__bf16)0.0f;
    __syncthreads();

    for (int t = 0; t < 512; ++t) {
        // ---- phase 1: gh = h @ w_hh^T (16 x 768), 8 waves x 6 N-tiles ----
        v8f acc[6] = {};
        for (int kk = 0; kk < 256; kk += 32) {
            v16bf a = frag_a(hb + kk, HSTRIDE);
#pragma unroll
            for (int nt = 0; nt < 6; ++nt) {
                int colb = (wave * 6 + nt) * 16;
                v16bf b  = frag_b(whh, 256, colb, kk);
                acc[nt]  = wmma_bf16(a, b, acc[nt]);
            }
        }
        {
            int hi = lane >> 4;
#pragma unroll
            for (int nt = 0; nt < 6; ++nt) {
                int col = (wave * 6 + nt) * 16 + (lane & 15);
                float av[8];
#pragma unroll
                for (int i = 0; i < 8; ++i) av[i] = acc[nt][i];
#pragma unroll
                for (int i = 0; i < 8; ++i)
                    gh[(i + 8 * hi) * 768 + col] = av[i];
            }
        }
        __syncthreads();

        // ---- phase 2: gates + state update ----
        {
            int bimg = tid >> 4;
            int j0   = (tid & 15) * 16;
            const float* gib = gi + ((size_t)bimg * 512 + t) * 768;
#pragma unroll 4
            for (int jj = 0; jj < 16; ++jj) {
                int j     = j0 + jj;
                float ghr = gh[bimg * 768 + j]       + bhh[j];
                float ghz = gh[bimg * 768 + 256 + j] + bhh[256 + j];
                float ghn = gh[bimg * 768 + 512 + j] + bhh[512 + j];
                float r   = 1.f / (1.f + __expf(-(gib[j] + ghr)));
                float z   = 1.f / (1.f + __expf(-(gib[256 + j] + ghz)));
                float nn  = tanhf(gib[512 + j] + r * ghn);
                float hn  = (1.f - z) * nn + z * hf[bimg * 256 + j];
                hf[bimg * 256 + j]     = hn;
                hb[bimg * HSTRIDE + j] = (__bf16)hn;
            }
        }
        __syncthreads();

        // ---- phase 3: out = h_new @ w_cls^T + b_cls (16x16), wave 0 ----
        if (wave == 0) {
            v8f cacc = {};
            for (int kk = 0; kk < 256; kk += 32) {
                v16bf a = frag_a(hb + kk, HSTRIDE);
                v16bf b = frag_b(wcls, 256, 0, kk);
                cacc    = wmma_bf16(a, b, cacc);
            }
            int hi     = lane >> 4;
            int nbc    = lane & 15;
            float bias = bcls[nbc];
            float av[8];
#pragma unroll
            for (int i = 0; i < 8; ++i) av[i] = cacc[i];
#pragma unroll
            for (int i = 0; i < 8; ++i) {
                int bimg = i + 8 * hi;
                outp[((size_t)bimg * 512 + t) * 16 + nbc] = av[i] + bias;
            }
        }
        __syncthreads();
    }
}

// ---------------------------------------------------------------------------
// Host launcher
// ---------------------------------------------------------------------------
extern "C" void kernel_launch(void* const* d_in, const int* in_sizes, int n_in,
                              void* d_out, int out_size, void* d_ws, size_t ws_size,
                              hipStream_t stream) {
    const float* x     = (const float*)d_in[0];
    const float* w1    = (const float*)d_in[1];
    const float* b1    = (const float*)d_in[2];
    const float* g1    = (const float*)d_in[3];
    const float* bt1   = (const float*)d_in[4];
    const float* m1    = (const float*)d_in[5];
    const float* v1    = (const float*)d_in[6];
    const float* w2    = (const float*)d_in[7];
    const float* b2    = (const float*)d_in[8];
    const float* g2    = (const float*)d_in[9];
    const float* bt2   = (const float*)d_in[10];
    const float* m2    = (const float*)d_in[11];
    const float* v2    = (const float*)d_in[12];
    const float* w3    = (const float*)d_in[13];
    const float* b3    = (const float*)d_in[14];
    const float* g3    = (const float*)d_in[15];
    const float* bt3   = (const float*)d_in[16];
    const float* m3    = (const float*)d_in[17];
    const float* v3    = (const float*)d_in[18];
    const float* w_ih  = (const float*)d_in[19];
    const float* w_hh  = (const float*)d_in[20];
    const float* b_ih  = (const float*)d_in[21];
    const float* b_hh  = (const float*)d_in[22];
    const float* w_cls = (const float*)d_in[23];
    const float* b_cls = (const float*)d_in[24];

    unsigned char* ws = (unsigned char*)d_ws;
    size_t off        = 0;
    auto alloc = [&](size_t bytes) -> void* {
        void* p = ws + off;
        off += (bytes + 255) & ~(size_t)255;
        return p;
    };

    __bf16* A0    = (__bf16*)alloc((size_t)16 * 512 * 8 * 256 * 2); // conv1 out
    __bf16* A1    = (__bf16*)alloc((size_t)16 * 512 * 2 * 256 * 2); // conv2 out
    __bf16* F     = (__bf16*)alloc((size_t)16 * 512 * 256 * 2);     // feats
    float*  GI    = (float*)alloc((size_t)16 * 512 * 768 * 4);      // gi_all
    __bf16* W2R   = (__bf16*)alloc((size_t)25 * 65536 * 2);
    __bf16* W3R   = (__bf16*)alloc((size_t)25 * 65536 * 2);
    __bf16* WIHB  = (__bf16*)alloc((size_t)768 * 256 * 2);
    __bf16* WHHB  = (__bf16*)alloc((size_t)768 * 256 * 2);
    __bf16* WCLSB = (__bf16*)alloc((size_t)16 * 256 * 2);
    (void)ws_size; (void)n_in; (void)in_sizes; (void)out_size;

    // Weight prep
    k_reorder_convw<<<(25 * 65536 + 255) / 256, 256, 0, stream>>>(w2, W2R);
    k_reorder_convw<<<(25 * 65536 + 255) / 256, 256, 0, stream>>>(w3, W3R);
    k_convert_bf16<<<(768 * 256 + 255) / 256, 256, 0, stream>>>(w_ih, WIHB, 768 * 256);
    k_convert_bf16<<<(768 * 256 + 255) / 256, 256, 0, stream>>>(w_hh, WHHB, 768 * 256);
    k_convert_bf16<<<(16 * 256 + 255) / 256, 256, 0, stream>>>(w_cls, WCLSB, 16 * 256);

    // conv1 fused (direct, Cin=1)
    k_conv1<<<16 * 512 * 8, 256, 0, stream>>>(x, w1, b1, g1, bt1, m1, v1, A0);

    // conv2: M=65536 -> 512 M-tiles x 2 N-tiles (128x128)
    k_conv_gemm<8, 4><<<512 * 2, 256, 0, stream>>>(A0, W2R, b2, g2, bt2, m2, v2, A1);

    // conv3: M=16384 -> 128 M-tiles x 2 N-tiles
    k_conv_gemm<2, 2><<<128 * 2, 256, 0, stream>>>(A1, W3R, b3, g3, bt3, m3, v3, F);

    // gi_all GEMM: M=8192 -> 64 M-tiles x 6 N-tiles (128x128)
    k_gemm_gi<<<64 * 6, 256, 0, stream>>>(F, WIHB, b_ih, GI);

    // GRU scan + classifier (single persistent workgroup)
    k_gru<<<1, 256, 0, stream>>>(GI, WHHB, b_hh, WCLSB, b_cls, (float*)d_out);
}
